// MoEMLP_2052994367552
// MI455X (gfx1250) — compile-verified, matrix operand order checked by reading, not run on hardware
//
#include <hip/hip_runtime.h>

// ---------------- problem constants ----------------
#define E_    8
#define KSEL  2
#define DDIM  2048
#define IDIM  5632
#define TTOK  4096          // B*S = 2*2048
#define CAP   1024          // T*K/E * CAP_FACTOR
#define TD    ((size_t)TTOK * DDIM)

// ---------------- workspace layout (bytes) ----------------
#define WS_SMA    0                    // float [TTOK*E_]          131072
#define WS_SEL    131072               // int   [TTOK*KSEL]         32768
#define WS_CMB    163840               // float [TTOK*KSEL]         32768
#define WS_ACC    196608               // float [32] counts8,probs8,z
#define WS_TOK    196736               // int   [E_*CAP]            32768
#define WS_SCALE  229504               // float [E_*CAP]            32768
#define WS_ACT    524288               // ushort(bf16) [E_*CAP*IDIM] ~92.3MB

typedef __attribute__((ext_vector_type(16))) __bf16 v16bf;
typedef __attribute__((ext_vector_type(2)))  __bf16 bf16x2;
typedef __attribute__((ext_vector_type(8)))  float  v8f;
typedef __attribute__((ext_vector_type(4)))  float  f32x4;   // clang vector (NT-load capable)

union Frag16 { v16bf v; unsigned u[8]; unsigned short s[16]; };

// fp32 -> packed bf16 pair via native converts (RNE in hardware)
__device__ __forceinline__ unsigned pack2(float a, float b) {
    bf16x2 p;
    p.x = (__bf16)a;
    p.y = (__bf16)b;
    return __builtin_bit_cast(unsigned, p);
}
__device__ __forceinline__ unsigned short f2bf(float f) {
    __bf16 h = (__bf16)f;
    return __builtin_bit_cast(unsigned short, h);
}
__device__ __forceinline__ float fast_rcp(float x) {
    return __builtin_amdgcn_rcpf(x);
}

// ---------------- utility: zero kernels ----------------
__global__ void moe_zero_y(float* __restrict__ y) {
    size_t i = ((size_t)blockIdx.x * blockDim.x + threadIdx.x) * 4;
    float4 z = {0.f, 0.f, 0.f, 0.f};
    *(float4*)(y + i) = z;
}
__global__ void moe_zero_acc(float* __restrict__ acc) {
    if (threadIdx.x < 32) acc[threadIdx.x] = 0.f;
}

// ---------------- router: logits, softmax stats, top-3, combine ----------------
__global__ void moe_router(const float* __restrict__ x, const float* __restrict__ r,
                           const float* __restrict__ rb,
                           float* __restrict__ s_ma, int* __restrict__ sel,
                           float* __restrict__ cmb, float* __restrict__ acc) {
    const int lane  = threadIdx.x & 31;
    const int token = blockIdx.x * 8 + (threadIdx.x >> 5);
    float a[E_] = {0.f, 0.f, 0.f, 0.f, 0.f, 0.f, 0.f, 0.f};

    const float* xp = x + (size_t)token * DDIM;
    for (int d = lane; d < DDIM; d += 32) {
        float xv = xp[d];
        const float4* rp = (const float4*)(r + (size_t)d * E_);
        float4 r0 = rp[0], r1 = rp[1];
        a[0] += xv * r0.x; a[1] += xv * r0.y; a[2] += xv * r0.z; a[3] += xv * r0.w;
        a[4] += xv * r1.x; a[5] += xv * r1.y; a[6] += xv * r1.z; a[7] += xv * r1.w;
    }
    #pragma unroll
    for (int off = 16; off; off >>= 1)
        #pragma unroll
        for (int e = 0; e < E_; e++) a[e] += __shfl_xor(a[e], off, 32);

    if (lane == 0) {
        float maxv = a[0];
        #pragma unroll
        for (int e = 1; e < E_; e++) maxv = fmaxf(maxv, a[e]);
        float p[E_], den = 0.f;
        #pragma unroll
        for (int e = 0; e < E_; e++) { p[e] = __expf(a[e] - maxv); den += p[e]; }
        float inv = fast_rcp(den);
        float lse = maxv + __logf(den);

        int   i0 = -1, i1 = -1;
        float v0 = -1e30f, v1 = -1e30f, v2 = -1e30f;
        #pragma unroll
        for (int e = 0; e < E_; e++) {
            float v = a[e] + rb[e];
            if (v > v0)      { v2 = v1; v1 = v0; i1 = i0; v0 = v; i0 = e; }
            else if (v > v1) { v2 = v1; v1 = v;  i1 = e; }
            else if (v > v2) { v2 = v; }
        }
        sel[token * 2 + 0] = i0;
        sel[token * 2 + 1] = i1;
        cmb[token * 2 + 0] = fast_rcp(1.f + __expf(-a[i0]));
        cmb[token * 2 + 1] = fast_rcp(1.f + __expf(-a[i1]));
        #pragma unroll
        for (int e = 0; e < E_; e++) s_ma[token * E_ + e] = a[e] - v2;

        atomicAdd(&acc[i0], 1.f);
        atomicAdd(&acc[i1], 1.f);
        #pragma unroll
        for (int e = 0; e < E_; e++) atomicAdd(&acc[8 + e], p[e] * inv);
        atomicAdd(&acc[16], lse * lse);
    }
}

// ---------------- qb_beta: per-expert 1024th-largest of s_ma column ----------------
__global__ void moe_qbbeta(const float* __restrict__ s_ma, float* __restrict__ out_qb) {
    const int e = blockIdx.x, tid = threadIdx.x;
    __shared__ unsigned keys[TTOK];
    __shared__ int cntS;
    for (int i = tid; i < TTOK; i += 256) {
        unsigned u = __float_as_uint(s_ma[(size_t)i * E_ + e]);
        keys[i] = (u & 0x80000000u) ? ~u : (u | 0x80000000u);
    }
    __syncthreads();
    unsigned prefix = 0;
    int k = TTOK * KSEL / E_;   // 1024
    for (int bit = 31; bit >= 0; --bit) {
        unsigned b = 1u << bit;
        unsigned hi = ~(b | (b - 1u));
        if (tid == 0) cntS = 0;
        __syncthreads();
        int c = 0;
        for (int i = tid; i < TTOK; i += 256) {
            unsigned key = keys[i];
            if (((key & hi) == prefix) && (key & b)) c++;
        }
        atomicAdd(&cntS, c);
        __syncthreads();
        int cnt = cntS;
        __syncthreads();
        if (cnt >= k) prefix |= b; else k -= cnt;
    }
    if (tid == 0) {
        unsigned u = (prefix & 0x80000000u) ? (prefix ^ 0x80000000u) : ~prefix;
        out_qb[e] = __uint_as_float(u);
    }
}

// ---------------- dispatch: flat-order positions, capacity drop, gather index ----------------
__global__ void moe_dispatch(const int* __restrict__ sel, const float* __restrict__ cmb,
                             int* __restrict__ tok_of, float* __restrict__ scale) {
    __shared__ int cnt[256][E_];
    const int tid = threadIdx.x;
    #pragma unroll
    for (int j = 0; j < 32; j++) { tok_of[tid * 32 + j] = -1; scale[tid * 32 + j] = 0.f; }
    int loc[E_] = {0, 0, 0, 0, 0, 0, 0, 0};
    const int base = tid * 32;
    for (int j = 0; j < 32; j++) loc[sel[base + j]]++;
    #pragma unroll
    for (int e = 0; e < E_; e++) cnt[tid][e] = loc[e];
    __syncthreads();
    if (tid == 0) {
        int run[E_] = {0, 0, 0, 0, 0, 0, 0, 0};
        for (int t = 0; t < 256; t++)
            for (int e = 0; e < E_; e++) { int v = cnt[t][e]; cnt[t][e] = run[e]; run[e] += v; }
    }
    __syncthreads();
    #pragma unroll
    for (int e = 0; e < E_; e++) loc[e] = cnt[tid][e];
    for (int j = 0; j < 32; j++) {
        int slot = base + j, e = sel[slot];
        int pos = loc[e]++;
        if (pos < CAP) {
            tok_of[e * CAP + pos] = slot >> 1;     // token = slot / K
            scale[e * CAP + pos]  = cmb[slot];
        }
    }
}

// ---------------- fragment index helper (ISA 16-bit A/B layout) ----------------
__device__ __forceinline__ int frag_k(int v, int kb) {
    return (v < 4) ? (kb + 2 * v) : (16 + kb + 2 * (v - 4));
}

// ---------------- GEMM1: buf(1024xD) * Wgu(Dx2I) -> SiLU(gate)*up -> act bf16 ----------------
// grid (I/64, CAP/128, E), block 256 (8 waves, 4x2 of 32x32 wave tiles), K-step 32,
// double-buffered LDS, B stored transposed [n][k] for ds_load_b128 fragment loads.
__global__ __launch_bounds__(256) void moe_gemm1(
    const float* __restrict__ x, const float* __restrict__ wgu,
    const int* __restrict__ tok_of, unsigned short* __restrict__ act) {
    const int e     = blockIdx.z;
    const int mBase = blockIdx.y * 128;
    const int nBase = blockIdx.x * 64;
    const int tid   = threadIdx.x;
    const int lane  = tid & 31, wid = tid >> 5;
    const int wm = (wid >> 1) * 32, wn = (wid & 1) * 32;

    __shared__ __align__(16) unsigned short As[2][128][40];
    __shared__ __align__(16) unsigned short Bg[2][64][40];   // transposed [n][k]
    __shared__ __align__(16) unsigned short Bu[2][64][40];   // transposed [n][k]

    // A staging: 2 threads per row, 16 K-elements each; branch-free via mask
    const int arow = tid >> 1;
    const int acol = (tid & 1) * 16;
    const int atok = tok_of[e * CAP + mBase + arow];
    const float amask = (atok >= 0) ? 1.f : 0.f;
    const float* xrow = x + (size_t)(atok >= 0 ? atok : 0) * DDIM;

    // B staging: 128 threads per matrix, each 2 K-rows x 8 N-cols, packed transpose
    const int bmat = tid >> 7;              // 0 = gate, 1 = up
    const int bu   = tid & 127;
    const int bkp  = (bu >> 3) * 2;         // k-pair base 0..30
    const int bng  = (bu & 7) * 8;          // n-group 0..56
    const float* bsrc = wgu + (size_t)e * DDIM * 2 * IDIM
                            + (size_t)bmat * IDIM + nBase + bng;

    auto stage = [&](int buf, int k0) {
        const f32x4* xp = (const f32x4*)(xrow + k0 + acol);
        f32x4 f0 = xp[0], f1 = xp[1], f2 = xp[2], f3 = xp[3];
        uint4 pa, pb;
        pa.x = pack2(f0.x * amask, f0.y * amask);
        pa.y = pack2(f0.z * amask, f0.w * amask);
        pa.z = pack2(f1.x * amask, f1.y * amask);
        pa.w = pack2(f1.z * amask, f1.w * amask);
        pb.x = pack2(f2.x * amask, f2.y * amask);
        pb.y = pack2(f2.z * amask, f2.w * amask);
        pb.z = pack2(f3.x * amask, f3.y * amask);
        pb.w = pack2(f3.z * amask, f3.w * amask);
        *(uint4*)&As[buf][arow][acol]     = pa;
        *(uint4*)&As[buf][arow][acol + 8] = pb;

        const f32x4* r0 = (const f32x4*)(bsrc + (size_t)(k0 + bkp) * 2 * IDIM);
        const f32x4* r1 = (const f32x4*)(bsrc + (size_t)(k0 + bkp + 1) * 2 * IDIM);
        f32x4 p0 = __builtin_nontemporal_load(r0);
        f32x4 p1 = __builtin_nontemporal_load(r0 + 1);
        f32x4 q0 = __builtin_nontemporal_load(r1);
        f32x4 q1 = __builtin_nontemporal_load(r1 + 1);
        float g0[8] = {p0.x, p0.y, p0.z, p0.w, p1.x, p1.y, p1.z, p1.w};
        float g1[8] = {q0.x, q0.y, q0.z, q0.w, q1.x, q1.y, q1.z, q1.w};
        unsigned short (*Bt)[40] = bmat ? Bu[buf] : Bg[buf];
        #pragma unroll
        for (int j = 0; j < 8; j++)
            *(unsigned*)&Bt[bng + j][bkp] = pack2(g0[j], g1[j]);
    };

    v8f accG[2][2] = {};
    v8f accU[2][2] = {};

    stage(0, 0);
    __syncthreads();
    const int NK = DDIM / 32;
    for (int kk = 0; kk < NK; kk++) {
        if (kk + 1 < NK) stage((kk + 1) & 1, (kk + 1) * 32);
        const int buf = kk & 1;

        Frag16 fa[2], fg[2], fu[2];
        const int rsel = lane & 15;
        const int kb   = (lane >> 4) * 8;
        #pragma unroll
        for (int mi = 0; mi < 2; mi++) {
            const int row = wm + mi * 16 + rsel;
            #pragma unroll
            for (int v = 0; v < 8; v++)
                fa[mi].u[v] = *(const unsigned*)&As[buf][row][frag_k(v, kb)];
        }
        #pragma unroll
        for (int ni = 0; ni < 2; ni++) {
            const int col = wn + ni * 16 + rsel;
            #pragma unroll
            for (int v = 0; v < 8; v++) {
                fg[ni].u[v] = *(const unsigned*)&Bg[buf][col][frag_k(v, kb)];
                fu[ni].u[v] = *(const unsigned*)&Bu[buf][col][frag_k(v, kb)];
            }
        }
        #pragma unroll
        for (int mi = 0; mi < 2; mi++)
            #pragma unroll
            for (int ni = 0; ni < 2; ni++) {
                accG[mi][ni] = __builtin_amdgcn_wmma_f32_16x16x32_bf16(
                    false, fa[mi].v, false, fg[ni].v, (short)0, accG[mi][ni], false, false);
                accU[mi][ni] = __builtin_amdgcn_wmma_f32_16x16x32_bf16(
                    false, fa[mi].v, false, fu[ni].v, (short)0, accU[mi][ni], false, false);
            }
        __syncthreads();
    }

    // epilogue: act = silu(gate) * up = g * rcp(1 + exp(-g)) * u, store bf16
    const int ncol = lane & 15;
    const int rofs = (lane >> 4) * 8;
    #pragma unroll
    for (int mi = 0; mi < 2; mi++)
        #pragma unroll
        for (int ni = 0; ni < 2; ni++)
            #pragma unroll
            for (int r = 0; r < 8; r++) {
                const int c = mBase + wm + mi * 16 + rofs + r;
                const int i = nBase + wn + ni * 16 + ncol;
                float g = accG[mi][ni][r];
                float u = accU[mi][ni][r];
                float s = g * fast_rcp(1.f + __expf(-g)) * u;
                act[(size_t)(e * CAP + c) * IDIM + i] = f2bf(s);
            }
}

// ---------------- GEMM2: act(1024xI) * Wdown(IxD) -> scaled scatter-add into y ----------------
// grid (D/128, CAP/128, E), block 256 (8 waves, 4x2 of 32x64 wave tiles), K-step 32,
// double-buffered LDS, transposed B tiles.
__global__ __launch_bounds__(256) void moe_gemm2(
    const unsigned short* __restrict__ act, const float* __restrict__ wd,
    const int* __restrict__ tok_of, const float* __restrict__ scale,
    float* __restrict__ y) {
    const int e     = blockIdx.z;
    const int mBase = blockIdx.y * 128;
    const int nBase = blockIdx.x * 128;
    const int tid   = threadIdx.x;
    const int lane  = tid & 31, wid = tid >> 5;
    const int wm = (wid >> 1) * 32, wn = (wid & 1) * 64;

    __shared__ __align__(16) unsigned short As[2][128][40];
    __shared__ __align__(16) unsigned short Bt[2][128][40];  // transposed [n][k]

    const int arow = tid >> 1;
    const int acol = (tid & 1) * 16;
    const unsigned short* ap = act + (size_t)(e * CAP + mBase + arow) * IDIM;

    const int bkp = (tid >> 4) * 2;         // k-pair base 0..30
    const int bng = (tid & 15) * 8;         // n-group 0..120
    const float* bsrc = wd + (size_t)e * IDIM * DDIM + nBase + bng;

    auto stage = [&](int buf, int k0) {
        const uint4* q = (const uint4*)(ap + k0 + acol);
        uint4 q0 = q[0], q1 = q[1];
        *(uint4*)&As[buf][arow][acol]     = q0;
        *(uint4*)&As[buf][arow][acol + 8] = q1;

        const f32x4* r0 = (const f32x4*)(bsrc + (size_t)(k0 + bkp) * DDIM);
        const f32x4* r1 = (const f32x4*)(bsrc + (size_t)(k0 + bkp + 1) * DDIM);
        f32x4 p0 = __builtin_nontemporal_load(r0);
        f32x4 p1 = __builtin_nontemporal_load(r0 + 1);
        f32x4 s0 = __builtin_nontemporal_load(r1);
        f32x4 s1 = __builtin_nontemporal_load(r1 + 1);
        float g0[8] = {p0.x, p0.y, p0.z, p0.w, p1.x, p1.y, p1.z, p1.w};
        float g1[8] = {s0.x, s0.y, s0.z, s0.w, s1.x, s1.y, s1.z, s1.w};
        #pragma unroll
        for (int j = 0; j < 8; j++)
            *(unsigned*)&Bt[buf][bng + j][bkp] = pack2(g0[j], g1[j]);
    };

    v8f acc[2][4] = {};

    stage(0, 0);
    __syncthreads();
    const int NK = IDIM / 32;
    for (int kk = 0; kk < NK; kk++) {
        if (kk + 1 < NK) stage((kk + 1) & 1, (kk + 1) * 32);
        const int buf = kk & 1;

        Frag16 fa[2], fb[4];
        const int rsel = lane & 15;
        const int kb   = (lane >> 4) * 8;
        #pragma unroll
        for (int mi = 0; mi < 2; mi++) {
            const int row = wm + mi * 16 + rsel;
            #pragma unroll
            for (int v = 0; v < 8; v++)
                fa[mi].u[v] = *(const unsigned*)&As[buf][row][frag_k(v, kb)];
        }
        #pragma unroll
        for (int ni = 0; ni < 4; ni++) {
            const int col = wn + ni * 16 + rsel;
            #pragma unroll
            for (int v = 0; v < 8; v++)
                fb[ni].u[v] = *(const unsigned*)&Bt[buf][col][frag_k(v, kb)];
        }
        #pragma unroll
        for (int mi = 0; mi < 2; mi++)
            #pragma unroll
            for (int ni = 0; ni < 4; ni++)
                acc[mi][ni] = __builtin_amdgcn_wmma_f32_16x16x32_bf16(
                    false, fa[mi].v, false, fb[ni].v, (short)0, acc[mi][ni], false, false);
        __syncthreads();
    }

    // epilogue: per-row combine scale, atomic scatter to tokens
    const int ncol = lane & 15;
    const int rofs = (lane >> 4) * 8;
    #pragma unroll
    for (int mi = 0; mi < 2; mi++)
        #pragma unroll
        for (int ni = 0; ni < 4; ni++)
            #pragma unroll
            for (int r = 0; r < 8; r++) {
                const int c = mBase + wm + mi * 16 + rofs + r;
                const int n = nBase + wn + ni * 16 + ncol;
                const int t = tok_of[e * CAP + c];
                if (t >= 0) {
                    float v = acc[mi][ni][r] * scale[e * CAP + c];
                    atomicAdd(y + (size_t)t * DDIM + n, v);
                }
            }
}

// ---------------- finalize scalar losses ----------------
__global__ void moe_finalize(const float* __restrict__ acc, float* __restrict__ out) {
    if (threadIdx.x == 0 && blockIdx.x == 0) {
        float c[E_], total = 0.f;
        #pragma unroll
        for (int e = 0; e < E_; e++) { c[e] = acc[e]; total += c[e]; }
        total = fmaxf(total, 1.f);
        float ent = 0.f, lb = 0.f;
        #pragma unroll
        for (int e = 0; e < E_; e++) {
            float f = c[e] / total;
            ent -= f * __logf(f + 1e-6f);
            lb  += f * (float)KSEL * (acc[8 + e] / (float)TTOK);
        }
        lb *= (float)E_;
        #pragma unroll
        for (int e = 0; e < E_; e++) out[TD + e] = c[e];
        out[TD + 8]  = ent;
        out[TD + 9]  = lb;
        out[TD + 10] = acc[16] / (float)TTOK;
    }
}

// ---------------- launch ----------------
extern "C" void kernel_launch(void* const* d_in, const int* in_sizes, int n_in,
                              void* d_out, int out_size, void* d_ws, size_t ws_size,
                              hipStream_t stream) {
    (void)in_sizes; (void)n_in; (void)out_size; (void)ws_size;
    const float* x   = (const float*)d_in[0];
    const float* rtr = (const float*)d_in[1];
    const float* rb  = (const float*)d_in[2];
    const float* wgu = (const float*)d_in[3];
    const float* wd  = (const float*)d_in[4];

    float* out = (float*)d_out;
    char*  ws  = (char*)d_ws;

    float*          s_ma   = (float*)(ws + WS_SMA);
    int*            sel    = (int*)  (ws + WS_SEL);
    float*          cmb    = (float*)(ws + WS_CMB);
    float*          acc    = (float*)(ws + WS_ACC);
    int*            tok_of = (int*)  (ws + WS_TOK);
    float*          scale  = (float*)(ws + WS_SCALE);
    unsigned short* act    = (unsigned short*)(ws + WS_ACT);

    moe_zero_acc<<<1, 32, 0, stream>>>(acc);
    moe_zero_y<<<TD / (256 * 4), 256, 0, stream>>>(out);
    moe_router<<<TTOK / 8, 256, 0, stream>>>(x, rtr, rb, s_ma, sel, cmb, acc);
    moe_qbbeta<<<E_, 256, 0, stream>>>(s_ma, out + TD + 11);
    moe_dispatch<<<1, 256, 0, stream>>>(sel, cmb, tok_of, scale);
    moe_gemm1<<<dim3(IDIM / 64, CAP / 128, E_), 256, 0, stream>>>(x, wgu, tok_of, act);
    moe_gemm2<<<dim3(DDIM / 128, CAP / 128, E_), 256, 0, stream>>>(act, wd, tok_of, scale, out);
    moe_finalize<<<1, 32, 0, stream>>>(acc, out);
}
